// Only_high_GCN_85856396247990
// MI455X (gfx1250) — matrix operand order repaired and checked by reading, not compiled, not run on hardware
//
#include <hip/hip_runtime.h>
#include <hip/hip_bf16.h>

typedef __attribute__((ext_vector_type(16))) _Float16 v16h;
typedef __attribute__((ext_vector_type(8)))  float    v8f;
typedef __attribute__((ext_vector_type(4)))  float    v4f;

#define BN_EPS 1e-5f

// ---------------- zero helpers ----------------------------------------------
__global__ void zero_f32_kernel(float* __restrict__ p, int n) {
  for (int i = blockIdx.x * blockDim.x + threadIdx.x; i < n;
       i += gridDim.x * blockDim.x) p[i] = 0.f;
}
__global__ void zero_i32_kernel(int* __restrict__ p, int n) {
  for (int i = blockIdx.x * blockDim.x + threadIdx.x; i < n;
       i += gridDim.x * blockDim.x) p[i] = 0;
}

// ---------------- coalesced column accumulation (sum / sumsq) ----------------
// 512 threads map onto consecutive columns of a row -> fully coalesced reads.
__global__ void __launch_bounds__(512) col_accum_kernel(
    const float* __restrict__ X, int N, int C, int log2C,
    float* __restrict__ sums, float* __restrict__ sumsq) {
  const int col  = threadIdx.x & (C - 1);
  const int rsub = threadIdx.x >> log2C;
  const int rowsPerIter = 512 >> log2C;
  float s = 0.f, q = 0.f;
  for (int r = blockIdx.x * rowsPerIter + rsub; r < N;
       r += gridDim.x * rowsPerIter) {
    float v = X[(size_t)r * C + col];
    s += v; q += v * v;
  }
  atomicAdd(&sums[col], s);
  atomicAdd(&sumsq[col], q);
}

__global__ void bn_finalize_kernel(const float* __restrict__ sums,
                                   const float* __restrict__ sumsq,
                                   const float* __restrict__ gamma,
                                   const float* __restrict__ beta,
                                   int N, int C,
                                   float* __restrict__ scale,
                                   float* __restrict__ shift) {
  int c = blockIdx.x * blockDim.x + threadIdx.x;
  if (c < C) {
    float mu  = sums[c] / (float)N;
    float var = sumsq[c] / (float)N - mu * mu;
    float rs  = rsqrtf(var + BN_EPS);
    float sc  = gamma[c] * rs;
    scale[c] = sc;
    shift[c] = beta[c] - mu * sc;
  }
}

// ---------------- pre-swizzle W (KxNcols, f32) into WMMA B-fragment order ----
// h=lane/16: elem e<8 = B[kt*32+8h+e][nt*16+lane%16]; e>=8: B[kt*32+16+8h+e-8][nt*16+lane%16]
__global__ void pack_b_kernel(const float* __restrict__ W, int K, int Ncols,
                              _Float16* __restrict__ out) {
  const int NT = Ncols / 16;
  const int total = (K / 32) * NT * 512;
  for (int idx = blockIdx.x * blockDim.x + threadIdx.x; idx < total;
       idx += gridDim.x * blockDim.x) {
    int e    = idx & 15;
    int lane = (idx >> 4) & 31;
    int nt   = (idx >> 9) % NT;
    int kt   = idx / (512 * NT);
    int k = kt * 32 + ((e < 8) ? ((lane >> 4) * 8 + e) : (16 + (lane >> 4) * 8 + (e - 8)));
    int n = nt * 16 + (lane & 15);
    out[idx] = (_Float16)W[(size_t)k * Ncols + n];
  }
}

// ---------------- fused BN-apply + WMMA GEMM ---------------------------------
// Out[N x 128] = act( BN(X[N x K]) @ W[K x 128] (+ bias) )
template <int K, bool BIAS_RELU>
__global__ void __launch_bounds__(256) gemm_bn_kernel(
    const float* __restrict__ X, const float* __restrict__ scale,
    const float* __restrict__ shift, const _Float16* __restrict__ Bpack,
    const float* __restrict__ bias, float* __restrict__ Out) {
  const int lane = threadIdx.x & 31;
  const int wave = threadIdx.x >> 5;
  const int tile = blockIdx.x * 8 + wave;   // one 16x16 D-tile per wave
  const int mt = tile >> 3;                 // row tile
  const int nt = tile & 7;                  // col tile (128/16 = 8)
  const int hs = (lane >> 4) * 8;           // K half-select per A layout
  const float* xrow = X + (size_t)(mt * 16 + (lane & 15)) * K;

  v8f c = {0.f, 0.f, 0.f, 0.f, 0.f, 0.f, 0.f, 0.f};
  for (int kt = 0; kt < K / 32; ++kt) {
    const int k0 = kt * 32;
    v16h a;
    {
      const int kk = k0 + hs;               // elems 0..7 : K = kk..kk+7
      v4f x0 = *(const v4f*)(xrow + kk);
      v4f x1 = *(const v4f*)(xrow + kk + 4);
      v4f s0 = *(const v4f*)(scale + kk);
      v4f s1 = *(const v4f*)(scale + kk + 4);
      v4f t0 = *(const v4f*)(shift + kk);
      v4f t1 = *(const v4f*)(shift + kk + 4);
#pragma unroll
      for (int j = 0; j < 4; ++j) a[j]     = (_Float16)(x0[j] * s0[j] + t0[j]);
#pragma unroll
      for (int j = 0; j < 4; ++j) a[4 + j] = (_Float16)(x1[j] * s1[j] + t1[j]);
    }
    {
      const int kk = k0 + 16 + hs;          // elems 8..15 : K = kk..kk+7
      v4f x0 = *(const v4f*)(xrow + kk);
      v4f x1 = *(const v4f*)(xrow + kk + 4);
      v4f s0 = *(const v4f*)(scale + kk);
      v4f s1 = *(const v4f*)(scale + kk + 4);
      v4f t0 = *(const v4f*)(shift + kk);
      v4f t1 = *(const v4f*)(shift + kk + 4);
#pragma unroll
      for (int j = 0; j < 4; ++j) a[8 + j]  = (_Float16)(x0[j] * s0[j] + t0[j]);
#pragma unroll
      for (int j = 0; j < 4; ++j) a[12 + j] = (_Float16)(x1[j] * s1[j] + t1[j]);
    }
    v16h b = *(const v16h*)(Bpack + ((size_t)(kt * 8 + nt) * 32 + lane) * 16);
    c = __builtin_amdgcn_wmma_f32_16x16x32_f16(false, a, false, b,
                                               (short)0, c, false, false);
  }
  const int n = nt * 16 + (lane & 15);
  const int rbase = mt * 16 + ((lane >> 4) * 8);
  if (BIAS_RELU) {
    const float bb = bias[n];
#pragma unroll
    for (int r = 0; r < 8; ++r) {
      float v = c[r] + bb;
      Out[(size_t)(rbase + r) * 128 + n] = v > 0.f ? v : 0.f;
    }
  } else {
#pragma unroll
    for (int r = 0; r < 8; ++r)
      Out[(size_t)(rbase + r) * 128 + n] = c[r];
  }
}

// ---------------- CSR construction -------------------------------------------
__global__ void deg_count_kernel(const int* __restrict__ dst, int E,
                                 int* __restrict__ cnt) {
  for (int e = blockIdx.x * blockDim.x + threadIdx.x; e < E;
       e += gridDim.x * blockDim.x)
    atomicAdd(&cnt[dst[e]], 1);
}

// single-block chunked Hillis-Steele exclusive scan; also emits dinv
__global__ void __launch_bounds__(1024) scan_dinv_kernel(
    const int* __restrict__ cnt, int N, int* __restrict__ offset,
    float* __restrict__ dinv) {
  __shared__ int buf[1024];
  __shared__ int carry;
  if (threadIdx.x == 0) carry = 0;
  __syncthreads();
  for (int base = 0; base < N; base += 1024) {
    int idx = base + (int)threadIdx.x;
    int v = (idx < N) ? cnt[idx] : 0;
    if (idx < N) dinv[idx] = rsqrtf((float)v + 1.0f);  // +1 self-loop
    buf[threadIdx.x] = v;
    __syncthreads();
    for (int o = 1; o < 1024; o <<= 1) {
      int t = (threadIdx.x >= (unsigned)o) ? buf[threadIdx.x - o] : 0;
      __syncthreads();
      buf[threadIdx.x] += t;
      __syncthreads();
    }
    if (idx < N) offset[idx] = carry + buf[threadIdx.x] - v;  // exclusive
    __syncthreads();
    if (threadIdx.x == 0) carry += buf[1023];
    __syncthreads();
  }
}

__global__ void fill_csr_kernel(const int* __restrict__ src,
                                const int* __restrict__ dst, int E,
                                const int* __restrict__ offset,
                                int* __restrict__ cursor,
                                int* __restrict__ adj) {
  for (int e = blockIdx.x * blockDim.x + threadIdx.x; e < E;
       e += gridDim.x * blockDim.x) {
    int d = dst[e];
    int pos = atomicAdd(&cursor[d], 1);
    adj[offset[d] + pos] = src[e];
  }
}

// ------- fused gather-aggregation + relu + classifier + log_softmax ----------
// one wave per destination row; no float atomics; H rows stream through L2
__global__ void __launch_bounds__(256) aggr_cls_kernel(
    const float* __restrict__ H, const float* __restrict__ dinv,
    const int* __restrict__ offset, const int* __restrict__ cnt,
    const int* __restrict__ adj, const float* __restrict__ bgcn,
    const float* __restrict__ Wcls, const float* __restrict__ bcls,
    float* __restrict__ logp) {
  __shared__ float sW[128 * 10];
  for (int i = threadIdx.x; i < 1280; i += blockDim.x) sW[i] = Wcls[i];
  __syncthreads();
  const int lane = threadIdx.x & 31;
  const int wave = threadIdx.x >> 5;
  const int row = blockIdx.x * 8 + wave;
  const float di = dinv[row];

  v4f acc = *(const v4f*)(bgcn + lane * 4);
  {
    v4f hv = *(const v4f*)(H + (size_t)row * 128 + lane * 4);
    float w = di * di;  // self-loop
#pragma unroll
    for (int t = 0; t < 4; ++t) acc[t] += hv[t] * w;
  }
  const int base = offset[row];
  const int n = cnt[row];
  for (int i = 0; i < n; ++i) {
    int s = adj[base + i];
    float w = dinv[s] * di;
    v4f hv = *(const v4f*)(H + (size_t)s * 128 + lane * 4);
#pragma unroll
    for (int t = 0; t < 4; ++t) acc[t] += hv[t] * w;
  }
#pragma unroll
  for (int t = 0; t < 4; ++t) acc[t] = acc[t] > 0.f ? acc[t] : 0.f;  // relu

  float z[10];
#pragma unroll
  for (int c = 0; c < 10; ++c) z[c] = 0.f;
#pragma unroll
  for (int t = 0; t < 4; ++t) {
    const float* wr = sW + (lane * 4 + t) * 10;
#pragma unroll
    for (int c = 0; c < 10; ++c) z[c] += acc[t] * wr[c];
  }
#pragma unroll
  for (int off = 16; off > 0; off >>= 1) {
#pragma unroll
    for (int c = 0; c < 10; ++c) z[c] += __shfl_xor(z[c], off, 32);
  }
  if (lane == 0) {
    float m = -3.402823466e+38f;
#pragma unroll
    for (int c = 0; c < 10; ++c) { z[c] += bcls[c]; m = fmaxf(m, z[c]); }
    float se = 0.f;
#pragma unroll
    for (int c = 0; c < 10; ++c) se += __expf(z[c] - m);
    float lse = __logf(se);
#pragma unroll
    for (int c = 0; c < 10; ++c)
      logp[(size_t)row * 10 + c] = z[c] - m - lse;
  }
}

// ---------------- host-side orchestration ------------------------------------
extern "C" void kernel_launch(void* const* d_in, const int* in_sizes, int n_in,
                              void* d_out, int out_size, void* d_ws, size_t ws_size,
                              hipStream_t stream) {
  const float* X_hi   = (const float*)d_in[0];
  const int*   edge   = (const int*)d_in[2];
  const float* gamma1 = (const float*)d_in[3];
  const float* beta1  = (const float*)d_in[4];
  const float* W_mlp  = (const float*)d_in[5];
  const float* b_mlp  = (const float*)d_in[6];
  const float* gamma2 = (const float*)d_in[7];
  const float* beta2  = (const float*)d_in[8];
  const float* W_gcn  = (const float*)d_in[9];
  const float* b_gcn  = (const float*)d_in[10];
  const float* W_cls  = (const float*)d_in[11];
  const float* b_cls  = (const float*)d_in[12];

  const int D_HI = 512, D_EMB = 128, D_HID = 128;
  const int N = in_sizes[0] / D_HI;
  const int E = in_sizes[2] / 2;
  const int* src = edge;
  const int* dst = edge + E;

  char* ws = (char*)d_ws;
  size_t off = 0;
  auto wsalloc = [&](size_t bytes) -> char* {
    char* p = ws + off;
    off = (off + bytes + 255) & ~(size_t)255;
    return p;
  };
  float*    sums   = (float*)wsalloc((size_t)D_HI * 4);
  float*    sumsq  = (float*)wsalloc((size_t)D_HI * 4);
  float*    scale1 = (float*)wsalloc((size_t)D_HI * 4);
  float*    shift1 = (float*)wsalloc((size_t)D_HI * 4);
  float*    scale2 = (float*)wsalloc((size_t)D_EMB * 4);
  float*    shift2 = (float*)wsalloc((size_t)D_EMB * 4);
  _Float16* Bp1    = (_Float16*)wsalloc((size_t)(D_HI / 32) * (D_EMB / 16) * 512 * 2);
  _Float16* Bp2    = (_Float16*)wsalloc((size_t)(D_EMB / 32) * (D_HID / 16) * 512 * 2);
  float*    Y      = (float*)wsalloc((size_t)N * D_EMB * 4);
  float*    H      = (float*)wsalloc((size_t)N * D_HID * 4);
  int*      cnt    = (int*)wsalloc((size_t)N * 4);
  int*      offs   = (int*)wsalloc((size_t)N * 4);
  int*      cursor = (int*)wsalloc((size_t)N * 4);
  float*    dinv   = (float*)wsalloc((size_t)N * 4);
  int*      adj    = (int*)wsalloc((size_t)E * 4);
  (void)ws_size; (void)n_in; (void)out_size;

  // 1) BN1 stats (coalesced accumulate + finalize)
  zero_f32_kernel<<<4, 256, 0, stream>>>(sums, D_HI);
  zero_f32_kernel<<<4, 256, 0, stream>>>(sumsq, D_HI);
  col_accum_kernel<<<512, 512, 0, stream>>>(X_hi, N, D_HI, 9, sums, sumsq);
  bn_finalize_kernel<<<2, 256, 0, stream>>>(sums, sumsq, gamma1, beta1, N, D_HI,
                                            scale1, shift1);
  // 2) pack weights for WMMA B-fragments
  pack_b_kernel<<<64, 256, 0, stream>>>(W_mlp, D_HI, D_EMB, Bp1);
  pack_b_kernel<<<16, 256, 0, stream>>>(W_gcn, D_EMB, D_HID, Bp2);
  // 3) Y = relu(BN1(X) @ W_mlp + b)          [WMMA f16]
  gemm_bn_kernel<512, true><<<N / 16, 256, 0, stream>>>(
      X_hi, scale1, shift1, Bp1, b_mlp, Y);
  // 4) BN2 stats over Y
  zero_f32_kernel<<<2, 256, 0, stream>>>(sums, D_EMB);
  zero_f32_kernel<<<2, 256, 0, stream>>>(sumsq, D_EMB);
  col_accum_kernel<<<512, 512, 0, stream>>>(Y, N, D_EMB, 7, sums, sumsq);
  bn_finalize_kernel<<<1, 256, 0, stream>>>(sums, sumsq, gamma2, beta2, N, D_EMB,
                                            scale2, shift2);
  // 5) H = BN2(Y) @ W_gcn                    [WMMA f16, BN fused]
  gemm_bn_kernel<128, false><<<N / 16, 256, 0, stream>>>(
      Y, scale2, shift2, Bp2, (const float*)nullptr, H);
  // 6) CSR build: counts -> exclusive scan (+dinv) -> bucket fill
  zero_i32_kernel<<<64, 256, 0, stream>>>(cnt, N);
  zero_i32_kernel<<<64, 256, 0, stream>>>(cursor, N);
  deg_count_kernel<<<2048, 256, 0, stream>>>(dst, E, cnt);
  scan_dinv_kernel<<<1, 1024, 0, stream>>>(cnt, N, offs, dinv);
  fill_csr_kernel<<<2048, 256, 0, stream>>>(src, dst, E, offs, cursor, adj);
  // 7) gather aggregation + relu + classifier + log_softmax (fused, no atomics)
  aggr_cls_kernel<<<N / 8, 256, 0, stream>>>(H, dinv, offs, cnt, adj, b_gcn,
                                             W_cls, b_cls, (float*)d_out);
}
